// GptModel_9929964388796
// MI455X (gfx1250) — matrix operand order repaired and checked by reading, not compiled, not run on hardware
//
#include <hip/hip_runtime.h>

// ---------------------------------------------------------------------------
// GPT single-layer forward for MI455X (gfx1250), wave32 + WMMA bf16.
// B=2, T=2048, D=1024, H=16, hd=64, V=32000.
// ---------------------------------------------------------------------------

typedef __bf16 v16bf __attribute__((ext_vector_type(16)));
typedef float  v8f   __attribute__((ext_vector_type(8)));

#define EMB  1024
#define VOC  32000
#define TSEQ 2048

union BFrag { v16bf v; uint4 q[2]; unsigned short u[16]; };

__device__ __forceinline__ unsigned short f2bf(float f) {
    return __builtin_bit_cast(unsigned short, (__bf16)f);   // native cvt
}

__device__ __forceinline__ uint4 pack8bf(float4 a, float4 b) {
    union { uint4 q; __bf16 h[8]; } u;
    u.h[0] = (__bf16)a.x; u.h[1] = (__bf16)a.y; u.h[2] = (__bf16)a.z; u.h[3] = (__bf16)a.w;
    u.h[4] = (__bf16)b.x; u.h[5] = (__bf16)b.y; u.h[6] = (__bf16)b.z; u.h[7] = (__bf16)b.w;
    return u.q;
}

__device__ __forceinline__ v8f wmma_bf16(v16bf a, v16bf b, v8f c) {
    return __builtin_amdgcn_wmma_f32_16x16x32_bf16(
        false, a, false, b, (short)0, c, false, false);
}

// pin: no instruction may be rescheduled across this point
__device__ __forceinline__ void sched_fence() { __builtin_amdgcn_sched_barrier(0); }

// ---------------------------------------------------------------------------
// 0a) f32 -> bf16 bulk convert (w_out), 8 elements/thread
// ---------------------------------------------------------------------------
__global__ __launch_bounds__(256) void k_cvt(const float* __restrict__ src,
                                             unsigned short* __restrict__ dst) {
    size_t i = (size_t)blockIdx.x * 256 + threadIdx.x;   // i-th group of 8
    const float4* s = (const float4*)(src + i * 8);
    ((uint4*)(dst + i * 8))[0] = pack8bf(s[0], s[1]);
}

// ---------------------------------------------------------------------------
// 0b) pack wq|wk|wv -> wqkv_bf [3072][1024] bf16
// ---------------------------------------------------------------------------
__global__ __launch_bounds__(256) void k_cvt_qkv(const float* __restrict__ wq,
                                                 const float* __restrict__ wk,
                                                 const float* __restrict__ wv,
                                                 unsigned short* __restrict__ dst) {
    size_t i = (size_t)blockIdx.x * 256 + threadIdx.x;   // group of 8 elements
    size_t e0 = i * 8;
    int row  = (int)(e0 >> 10);            // 0..3071
    int off  = (int)(e0 & 1023);
    int sect = row >> 10;
    int nloc = row & 1023;
    const float* W = (sect == 0) ? wq : ((sect == 1) ? wk : wv);
    const float4* s = (const float4*)(W + (size_t)nloc * EMB + off);
    ((uint4*)(dst + e0))[0] = pack8bf(s[0], s[1]);
}

// ---------------------------------------------------------------------------
// 1) embedding: x = tok_emb[idx] + pos ; emit f32 (residual) + bf16 (WMMA A)
// ---------------------------------------------------------------------------
__global__ __launch_bounds__(256) void k_embed(const int* __restrict__ xin,
                                               const float* __restrict__ tok,
                                               const float* __restrict__ pos,
                                               float* __restrict__ xf,
                                               unsigned short* __restrict__ xbf) {
    int row = blockIdx.x;            // 0..4095
    int t   = row & (TSEQ - 1);
    int tid = threadIdx.x;           // 256 threads -> 4 floats each
    int tk  = xin[row];
    float4 a = ((const float4*)(tok + (size_t)tk * EMB))[tid];
    float4 p = ((const float4*)(pos + (size_t)t  * EMB))[tid];
    float4 s; s.x = a.x + p.x; s.y = a.y + p.y; s.z = a.z + p.z; s.w = a.w + p.w;
    ((float4*)(xf + (size_t)row * EMB))[tid] = s;
    ushort4 h; h.x = f2bf(s.x); h.y = f2bf(s.y); h.z = f2bf(s.z); h.w = f2bf(s.w);
    ((ushort4*)(xbf + (size_t)row * EMB))[tid] = h;
}

// ---------------------------------------------------------------------------
// 2) QKV projection: [4096x1024] x [1024x3072] via WMMA bf16.
//    Double-buffered fragment sets, pinned with sched barriers so next-step
//    loads stay in flight across the current step's WMMAs (partial loadcnt
//    waits instead of full drains).
//    q,k -> qk_bf [4096][2048] ; v -> vT[(b*16+h)*64+e][2048]
// ---------------------------------------------------------------------------
struct FragSetQ { BFrag A; BFrag B[4]; };

__device__ __forceinline__ void qkv_load(FragSetQ& s,
                                         const unsigned short* ap0,
                                         const unsigned short* bp0,
                                         int k0, int kbA) {
    s.A.q[0] = *(const uint4*)(ap0 + k0 + kbA);
    s.A.q[1] = *(const uint4*)(ap0 + k0 + kbA + 8);
#pragma unroll
    for (int nt = 0; nt < 4; ++nt) {
        const unsigned short* wp = bp0 + (size_t)(nt * 16) * EMB + k0;
        s.B[nt].q[0] = *(const uint4*)(wp);
        s.B[nt].q[1] = *(const uint4*)(wp + 8);
    }
}

__device__ __forceinline__ void qkv_mma(const FragSetQ& s, v8f acc[4]) {
#pragma unroll
    for (int nt = 0; nt < 4; ++nt)
        acc[nt] = wmma_bf16(s.A.v, s.B[nt].v, acc[nt]);
}

__global__ __launch_bounds__(256) void k_qkv(const unsigned short* __restrict__ xbf,
                                             const unsigned short* __restrict__ wqkv,
                                             unsigned short* __restrict__ qk,
                                             unsigned short* __restrict__ vT) {
    int lane  = threadIdx.x & 31;
    int wid   = blockIdx.x * 8 + (threadIdx.x >> 5);
    int mtile = wid / 48;            // 256 row tiles
    int g     = wid % 48;            // 48 groups of 64 output cols
    int n0    = g * 64;              // global col in [0,3072)

    int col = lane & 15, rb = (lane >> 4) * 8;
    int kbA = (lane < 16) ? 0 : 16;
    int kbB = (lane < 16) ? 0 : 16;
    int mrow = mtile * 16 + (lane & 15);

    v8f acc[4] = {};
    const unsigned short* ap0 = xbf + (size_t)mrow * EMB;
    const unsigned short* bp0 = wqkv + (size_t)(n0 + col) * EMB + kbB;

    FragSetQ sA, sB;
    qkv_load(sA, ap0, bp0, 0, kbA);
    sched_fence();
    for (int i = 0; i < 16; ++i) {                 // 32 K-steps, 2 per iter
        qkv_load(sB, ap0, bp0, i * 64 + 32, kbA);
        sched_fence();
        qkv_mma(sA, acc);
        sched_fence();
        if (i < 15) qkv_load(sA, ap0, bp0, i * 64 + 64, kbA);
        sched_fence();
        qkv_mma(sB, acc);
        sched_fence();
    }

    int sect = n0 >> 10;             // 0=q 1=k 2=v (group of 64 never straddles)
#pragma unroll
    for (int nt = 0; nt < 4; ++nt) {
        int n = n0 + nt * 16 + col;  // global col 0..3071
#pragma unroll
        for (int j = 0; j < 8; ++j) {
            int row = mtile * 16 + rb + j;
            unsigned short hv = f2bf(acc[nt][j]);
            if (sect < 2) {
                qk[(size_t)row * 2048 + n] = hv;     // n < 2048 here
            } else {
                int f = n - 2048, h = f >> 6, e = f & 63;
                int bb = row >> 11, tt = row & (TSEQ - 1);
                vT[(size_t)((bb * 16 + h) * 64 + e) * TSEQ + tt] = hv;
            }
        }
    }
}

// ---------------------------------------------------------------------------
// 3) attention: flash-style streaming softmax, 32 keys per step.
//    V-fragment global loads issued right after the Q@K^T WMMAs so their
//    latency hides behind the LDS softmax work.
// ---------------------------------------------------------------------------
__global__ __launch_bounds__(256) void k_attn(const unsigned short* __restrict__ qk,
                                              const unsigned short* __restrict__ vT,
                                              float* __restrict__ y) {
    __shared__ __align__(16) float          sS[8][16 * 32];
    __shared__ __align__(16) unsigned short sP[8][16 * 32];
    __shared__ float                        sRow[8][16];

    int w    = threadIdx.x >> 5;
    int lane = threadIdx.x & 31;
    int wid  = blockIdx.x * 8 + w;
    int qt   = wid & 127;            // 128 query tiles of 16
    int bh   = wid >> 7;             // b*16 + h
    int h    = bh & 15, b = bh >> 4;

    int col = lane & 15, rb = (lane >> 4) * 8;
    int kbA = (lane < 16) ? 0 : 16;
    int kbB = (lane < 16) ? 0 : 16;
    int r   = lane & 15;             // softmax row handled by this lane
    int c0  = (lane >> 4) * 16;      // softmax col half

    // Q A-fragments (resident for the whole key loop)
    int mrow = b * TSEQ + qt * 16 + (lane & 15);
    const unsigned short* qp = qk + (size_t)mrow * 2048 + h * 64;
    BFrag aq0, aq1;
    aq0.q[0] = *(const uint4*)(qp + kbA);       aq0.q[1] = *(const uint4*)(qp + kbA + 8);
    aq1.q[0] = *(const uint4*)(qp + 32 + kbA);  aq1.q[1] = *(const uint4*)(qp + 32 + kbA + 8);

    float m_r = -1e30f, l_r = 0.0f;
    v8f acc[4] = {};

    for (int kb = 0; kb < TSEQ; kb += 32) {
        // ---- S tile (16 queries x 32 keys), scaled by 1/sqrt(64) ----
#pragma unroll
        for (int nt = 0; nt < 2; ++nt) {
            int key = b * TSEQ + kb + nt * 16 + col;
            const unsigned short* kp = qk + (size_t)key * 2048 + 1024 + h * 64;
            BFrag b0, b1;
            b0.q[0] = *(const uint4*)(kp + kbB);      b0.q[1] = *(const uint4*)(kp + kbB + 8);
            b1.q[0] = *(const uint4*)(kp + 32 + kbB); b1.q[1] = *(const uint4*)(kp + 32 + kbB + 8);
            v8f s = {};
            s = wmma_bf16(aq0.v, b0.v, s);
            s = wmma_bf16(aq1.v, b1.v, s);
#pragma unroll
            for (int j = 0; j < 8; ++j)
                sS[w][(rb + j) * 32 + nt * 16 + col] = s[j] * 0.125f;
        }

        // ---- issue V B-fragment loads early (independent of softmax) ----
        BFrag bv[4];
#pragma unroll
        for (int et = 0; et < 4; ++et) {
            const unsigned short* vp =
                vT + (size_t)(bh * 64 + et * 16 + col) * TSEQ + kb + kbB;
            bv[et].q[0] = *(const uint4*)vp;
            bv[et].q[1] = *(const uint4*)(vp + 8);
        }
        sched_fence();   // keep V loads in flight during softmax below
        __builtin_amdgcn_wave_barrier();

        // ---- online softmax update (lane handles row r, cols c0..c0+15) ----
        float vmax = -1e30f;
#pragma unroll
        for (int i = 0; i < 16; ++i)
            vmax = fmaxf(vmax, sS[w][r * 32 + c0 + i]);
        vmax = fmaxf(vmax, __shfl_xor(vmax, 16));
        float m_new = fmaxf(m_r, vmax);
        float corr  = __expf(m_r - m_new);
        float ps = 0.0f;
#pragma unroll
        for (int i = 0; i < 16; ++i) {
            float p = __expf(sS[w][r * 32 + c0 + i] - m_new);
            ps += p;
            sP[w][r * 32 + c0 + i] = f2bf(p);
        }
        ps += __shfl_xor(ps, 16);
        l_r = l_r * corr + ps;
        m_r = m_new;
        if (lane < 16) sRow[w][r] = corr;
        __builtin_amdgcn_wave_barrier();

        // rescale running Y accumulators by per-row correction
        float cs[8];
#pragma unroll
        for (int j = 0; j < 8; ++j) cs[j] = sRow[w][rb + j];
#pragma unroll
        for (int nt = 0; nt < 4; ++nt)
#pragma unroll
            for (int j = 0; j < 8; ++j) acc[nt][j] *= cs[j];

        // ---- Y += P @ V ----
        BFrag ap;
        const unsigned short* pp = &sP[w][r * 32];
        ap.q[0] = *(const uint4*)(pp + kbA);
        ap.q[1] = *(const uint4*)(pp + kbA + 8);
#pragma unroll
        for (int et = 0; et < 4; ++et)
            acc[et] = wmma_bf16(ap.v, bv[et].v, acc[et]);
    }

    // ---- finalize: divide by l, write y[b,t, h*64+e] ----
    if (lane < 16) sRow[w][r] = 1.0f / l_r;
    __builtin_amdgcn_wave_barrier();
    float cs[8];
#pragma unroll
    for (int j = 0; j < 8; ++j) cs[j] = sRow[w][rb + j];
#pragma unroll
    for (int et = 0; et < 4; ++et)
#pragma unroll
        for (int j = 0; j < 8; ++j)
            y[(size_t)(b * TSEQ + qt * 16 + rb + j) * EMB + h * 64 + et * 16 + col] =
                acc[et][j] * cs[j];
}

// ---------------------------------------------------------------------------
// 4) residual + LayerNorm -> bf16
// ---------------------------------------------------------------------------
__global__ __launch_bounds__(256) void k_ln(const float* __restrict__ x,
                                            const float* __restrict__ y,
                                            const float* __restrict__ g,
                                            const float* __restrict__ bta,
                                            unsigned short* __restrict__ out) {
    __shared__ float s1[256], s2[256];
    int row = blockIdx.x, tid = threadIdx.x;
    float4 xa = ((const float4*)(x + (size_t)row * EMB))[tid];
    float4 ya = ((const float4*)(y + (size_t)row * EMB))[tid];
    float z0 = xa.x + ya.x, z1 = xa.y + ya.y, z2 = xa.z + ya.z, z3 = xa.w + ya.w;
    s1[tid] = z0 + z1 + z2 + z3;
    s2[tid] = z0 * z0 + z1 * z1 + z2 * z2 + z3 * z3;
    __syncthreads();
    for (int s = 128; s > 0; s >>= 1) {
        if (tid < s) { s1[tid] += s1[tid + s]; s2[tid] += s2[tid + s]; }
        __syncthreads();
    }
    float mu  = s1[0] * (1.0f / EMB);
    float var = s2[0] * (1.0f / EMB) - mu * mu;
    float inv = rsqrtf(var + 1e-5f);
    float4 ga = ((const float4*)g)[tid];
    float4 ba = ((const float4*)bta)[tid];
    ushort4 hv;
    hv.x = f2bf((z0 - mu) * inv * ga.x + ba.x);
    hv.y = f2bf((z1 - mu) * inv * ga.y + ba.y);
    hv.z = f2bf((z2 - mu) * inv * ga.z + ba.z);
    hv.w = f2bf((z3 - mu) * inv * ga.w + ba.w);
    ((ushort4*)(out + (size_t)row * EMB))[tid] = hv;
}

// ---------------------------------------------------------------------------
// 5) logits = r @ w_out^T + b_out : [4096x1024]x[1024x32000] WMMA bf16.
//    Each wave: 32x64 output (2 M-tiles x 4 N-tiles, 8 accumulators).
//    Double-buffered fragment sets, pinned with sched barriers.
// ---------------------------------------------------------------------------
struct FragSetL { BFrag A0, A1; BFrag B[4]; };

__device__ __forceinline__ void lg_load(FragSetL& s,
                                        const unsigned short* ap0,
                                        const unsigned short* ap1,
                                        const unsigned short* bp0,
                                        int k0, int kbA) {
    s.A0.q[0] = *(const uint4*)(ap0 + k0 + kbA);
    s.A0.q[1] = *(const uint4*)(ap0 + k0 + kbA + 8);
    s.A1.q[0] = *(const uint4*)(ap1 + k0 + kbA);
    s.A1.q[1] = *(const uint4*)(ap1 + k0 + kbA + 8);
#pragma unroll
    for (int nt = 0; nt < 4; ++nt) {
        const unsigned short* wp = bp0 + (size_t)(nt * 16) * EMB + k0;
        s.B[nt].q[0] = *(const uint4*)(wp);
        s.B[nt].q[1] = *(const uint4*)(wp + 8);
    }
}

__device__ __forceinline__ void lg_mma(const FragSetL& s, v8f acc[2][4]) {
#pragma unroll
    for (int nt = 0; nt < 4; ++nt) {
        acc[0][nt] = wmma_bf16(s.A0.v, s.B[nt].v, acc[0][nt]);
        acc[1][nt] = wmma_bf16(s.A1.v, s.B[nt].v, acc[1][nt]);
    }
}

__global__ __launch_bounds__(256) void k_logits(const unsigned short* __restrict__ rbf,
                                                const unsigned short* __restrict__ wbf,
                                                const float* __restrict__ bout,
                                                float* __restrict__ out) {
    int lane = threadIdx.x & 31;
    int wid  = blockIdx.x * 8 + (threadIdx.x >> 5);
    int mgrp = wid / 500;            // 128 groups of 32 rows
    int ngrp = wid % 500;            // 500 groups of 64 vocab cols
    int n0   = ngrp * 64;

    int col = lane & 15, rb = (lane >> 4) * 8;
    int kbA = (lane < 16) ? 0 : 16;
    int kbB = (lane < 16) ? 0 : 16;

    v8f acc[2][4] = {};
    const unsigned short* ap0 = rbf + (size_t)(mgrp * 32 + (lane & 15)) * EMB;
    const unsigned short* ap1 = ap0 + (size_t)16 * EMB;
    const unsigned short* bp0 = wbf + (size_t)(n0 + col) * EMB + kbB;

    FragSetL sA, sB;
    lg_load(sA, ap0, ap1, bp0, 0, kbA);
    sched_fence();
    for (int i = 0; i < 16; ++i) {                 // 32 K-steps, 2 per iter
        lg_load(sB, ap0, ap1, bp0, i * 64 + 32, kbA);
        sched_fence();
        lg_mma(sA, acc);
        sched_fence();
        if (i < 15) lg_load(sA, ap0, ap1, bp0, i * 64 + 64, kbA);
        sched_fence();
        lg_mma(sB, acc);
        sched_fence();
    }

#pragma unroll
    for (int mt = 0; mt < 2; ++mt) {
#pragma unroll
        for (int nt = 0; nt < 4; ++nt) {
            int n = n0 + nt * 16 + col;
            float bo = bout[n];
#pragma unroll
            for (int j = 0; j < 8; ++j)
                out[(size_t)(mgrp * 32 + mt * 16 + rb + j) * VOC + n] =
                    acc[mt][nt][j] + bo;
        }
    }
}

// ---------------------------------------------------------------------------
// launcher — workspace layout (bytes):
//   x_f32    16777216  @ 0
//   x_bf16    8388608  @ 16777216
//   qk_bf16  16777216  @ 25165824
//   vT_bf16   8388608  @ 41943040
//   y_f32    16777216  @ 50331648
//   r_bf16    8388608  @ 67108864
//   wqkv_bf   6291456  @ 75497472
//   wout_bf  65536000  @ 81788928   (end 147324928, ~140.5 MB)
// ---------------------------------------------------------------------------
extern "C" void kernel_launch(void* const* d_in, const int* in_sizes, int n_in,
                              void* d_out, int out_size, void* d_ws, size_t ws_size,
                              hipStream_t stream) {
    const int*   xin  = (const int*)d_in[0];
    const float* tok  = (const float*)d_in[1];
    const float* pos  = (const float*)d_in[2];
    const float* wq   = (const float*)d_in[3];
    const float* wk   = (const float*)d_in[4];
    const float* wv   = (const float*)d_in[5];
    const float* lng  = (const float*)d_in[6];
    const float* lnb  = (const float*)d_in[7];
    const float* wout = (const float*)d_in[8];
    const float* bout = (const float*)d_in[9];

    char* ws = (char*)d_ws;
    float*          xf   = (float*)(ws + 0);
    unsigned short* xbf  = (unsigned short*)(ws + 16777216);
    unsigned short* qk   = (unsigned short*)(ws + 25165824);
    unsigned short* vT   = (unsigned short*)(ws + 41943040);
    float*          y    = (float*)(ws + 50331648);
    unsigned short* rbf  = (unsigned short*)(ws + 67108864);
    unsigned short* wqkv = (unsigned short*)(ws + 75497472);
    unsigned short* wbf  = (unsigned short*)(ws + 81788928);

    k_cvt    <<<16000, 256, 0, stream>>>(wout, wbf);          // 32000*1024/8/256
    k_cvt_qkv<<<1536,  256, 0, stream>>>(wq, wk, wv, wqkv);   // 3072*1024/8/256
    k_embed  <<<4096,  256, 0, stream>>>(xin, tok, pos, xf, xbf);
    k_qkv    <<<1536,  256, 0, stream>>>(xbf, wqkv, qk, vT);  // 256*48 waves
    k_attn   <<<512,   256, 0, stream>>>(qk, vT, y);          // 2*16*128 waves
    k_ln     <<<4096,  256, 0, stream>>>(xf, y, lng, lnb, rbf);
    k_logits <<<8000,  256, 0, stream>>>(rbf, wbf, bout, (float*)d_out); // 128*500 waves
}
